// FrameTransformerBlock_79783312490670
// MI455X (gfx1250) — compile-verified, matrix operand order checked by reading, not run on hardware
//
#include <hip/hip_runtime.h>
#include <hip/hip_bf16.h>

// ---------------- problem constants ----------------
constexpr int nB    = 4;
constexpr int nCH   = 8;
constexpr int nBINS = 1024;
constexpr int nW    = 512;
constexpr int nH    = 4;
constexpr int nD    = nBINS / nH;     // 256
constexpr int nFF   = 2048;
constexpr int nC2   = 2 * nFF;        // 4096
constexpr int nBW   = nB * nW;        // 2048 rows

// ---------------- WMMA types ----------------
typedef __attribute__((ext_vector_type(16))) __bf16 v16bf;
typedef __attribute__((ext_vector_type(8)))  float  v8f;

#define GEMM_BM 128
#define GEMM_BN 64
#define GEMM_BK 32

// ---------------- Tensor Data Mover (gfx1250) ----------------
#if defined(__gfx1250__) && __has_builtin(__builtin_amdgcn_tensor_load_to_lds) && \
    __has_builtin(__builtin_amdgcn_s_wait_tensorcnt)
#define USE_TDM 1
#endif

#ifdef USE_TDM
typedef __attribute__((ext_vector_type(4))) unsigned tdm_u4;
typedef __attribute__((ext_vector_type(8))) int      tdm_i8;
typedef __attribute__((ext_vector_type(4))) int      tdm_i4;

// 2D bf16 tile load: tile_rows x tile_k elements, row stride = stride_elems.
// LDS layout produced: rows packed consecutively, tile_k elems per row.
__device__ __forceinline__ void tdm_load_bf16_2d(const __bf16* gp, void* lp,
                                                 unsigned tile_k, unsigned tile_rows,
                                                 unsigned tensor_k, unsigned tensor_rows,
                                                 unsigned stride_elems)
{
  const unsigned long long ga = (unsigned long long)gp;
  const unsigned lds =
      (unsigned)(size_t)(__attribute__((address_space(3))) void*)lp;
  tdm_u4 g0 = (tdm_u4)0;
  g0[0] = 1u;                                            // count=1 (user D#)
  g0[1] = lds;                                           // lds_addr
  g0[2] = (unsigned)ga;                                  // global_addr[31:0]
  g0[3] = (unsigned)((ga >> 32) & 0x1FFFFFFu) | (2u << 30); // addr[56:32] | type=2
  tdm_i8 g1 = (tdm_i8)0;
  g1[0] = (int)(1u << 16);                               // wg_mask=0, data_size=1 (2B)
  g1[1] = (int)((tensor_k & 0xFFFFu) << 16);             // tensor_dim0[15:0]
  g1[2] = (int)(((tensor_k >> 16) & 0xFFFFu) |
                ((tensor_rows & 0xFFFFu) << 16));        // dim0 hi | dim1 lo
  g1[3] = (int)(((tensor_rows >> 16) & 0xFFFFu) |
                ((tile_k & 0xFFFFu) << 16));             // dim1 hi | tile_dim0
  g1[4] = (int)(tile_rows & 0xFFFFu);                    // tile_dim1 | tile_dim2=0
  g1[5] = (int)stride_elems;                             // tensor_dim0_stride[31:0]
  tdm_i4 z4 = (tdm_i4)0;
#if __clang_major__ >= 23
  tdm_i8 z8 = (tdm_i8)0;
  __builtin_amdgcn_tensor_load_to_lds(g0, g1, z4, z4, z8, 0);
#else
  __builtin_amdgcn_tensor_load_to_lds(g0, g1, z4, z4, 0);
#endif
}
#endif  // USE_TDM

// =====================================================================
// Batched NT GEMM: C[z] = A[z] (MxK, lda) * B[z]^T (B stored [N,K], ldb) + bias
// bf16 inputs, f32 accumulate via v_wmma_f32_16x16x32_bf16.
// Requires M%128==0, N%64==0, K%32==0 (true for every call in this net).
// Tile staging is done by the Tensor Data Mover (wave 0 issues two 2D tile
// descriptors, waits tensorcnt, workgroup barrier publishes LDS).
// =====================================================================
__global__ __launch_bounds__(256)
void k_gemm(const __bf16* __restrict__ A, long long sA, int lda,
            const __bf16* __restrict__ Bm, long long sB, int ldb,
            float* __restrict__ C, long long sC, int ldc,
            const float* __restrict__ bias,
            int M, int N, int K)
{
  __shared__ __align__(16) __bf16 As[GEMM_BM * GEMM_BK];
  __shared__ __align__(16) __bf16 Bs[GEMM_BN * GEMM_BK];

  const int z = blockIdx.z;
  A  += (long long)z * sA;
  Bm += (long long)z * sB;
  C  += (long long)z * sC;

  const int m0  = blockIdx.y * GEMM_BM;
  const int n0  = blockIdx.x * GEMM_BN;
  const int tid = threadIdx.x;
  const int lane = tid & 31;
  const int wv   = tid >> 5;
  const int wm   = (wv & 3) * 32;   // wave M offset inside tile
  const int wn   = (wv >> 2) * 32;  // wave N offset inside tile

  v8f acc[2][2] = {};               // zero-initialized f32 accumulators

  const int kb = (lane >> 4) * 8;   // per-lane K base (0 or 8) per ISA layout
  const int h4 = kb >> 3;           // uint4 index inside a 32-elem LDS row

  for (int k0 = 0; k0 < K; k0 += GEMM_BK) {
    __syncthreads();
#ifdef USE_TDM
    if (tid < 32) {
      tdm_load_bf16_2d(A + (long long)m0 * lda + k0, (void*)As,
                       GEMM_BK, GEMM_BM, (unsigned)lda, (unsigned)M, (unsigned)lda);
      tdm_load_bf16_2d(Bm + (long long)n0 * ldb + k0, (void*)Bs,
                       GEMM_BK, GEMM_BN, (unsigned)ldb, (unsigned)N, (unsigned)ldb);
      __builtin_amdgcn_s_wait_tensorcnt(0);
    }
#else
    {
      const int r = tid >> 1, half = tid & 1;
      const uint4* src = (const uint4*)(A + (long long)(m0 + r) * lda + k0 + half * 16);
      uint4* dst = (uint4*)(As + r * GEMM_BK + half * 16);
      dst[0] = src[0];
      dst[1] = src[1];
    }
    if (tid < 128) {
      const int r = tid >> 1, half = tid & 1;
      const uint4* src = (const uint4*)(Bm + (long long)(n0 + r) * ldb + k0 + half * 16);
      uint4* dst = (uint4*)(Bs + r * GEMM_BK + half * 16);
      dst[0] = src[0];
      dst[1] = src[1];
    }
#endif
    // speculative prefetch of next K tile (global_prefetch_b8)
    if (k0 + GEMM_BK < K) {
      const int r = tid >> 1;
      __builtin_prefetch(A + (long long)(m0 + r) * lda + k0 + GEMM_BK, 0, 1);
    }
    __syncthreads();

    // ---- build fragments (two ds_load_b128 each, ISA 16-bit layout) ----
    union FB { v16bf v; uint4 q[2]; };
    FB a[2], b[2];
#pragma unroll
    for (int i = 0; i < 2; ++i) {
      const uint4* pa = (const uint4*)(As + (wm + i * 16 + (lane & 15)) * GEMM_BK);
      a[i].q[0] = pa[h4];
      a[i].q[1] = pa[h4 + 2];
      const uint4* pb = (const uint4*)(Bs + (wn + i * 16 + (lane & 15)) * GEMM_BK);
      b[i].q[0] = pb[h4];
      b[i].q[1] = pb[h4 + 2];
    }
#pragma unroll
    for (int i = 0; i < 2; ++i)
#pragma unroll
      for (int j = 0; j < 2; ++j)
        acc[i][j] = __builtin_amdgcn_wmma_f32_16x16x32_bf16(
            false, a[i].v, false, b[j].v, (short)0, acc[i][j], false, false);
  }

  // ---- store (C/D layout: VGPR r -> M = r + 8*(lane>=16), N = lane&15) ----
#pragma unroll
  for (int i = 0; i < 2; ++i) {
#pragma unroll
    for (int j = 0; j < 2; ++j) {
      const int col = n0 + wn + j * 16 + (lane & 15);
      const float bv = bias ? bias[col] : 0.0f;
#pragma unroll
      for (int r = 0; r < 8; ++r) {
        const int row = m0 + wm + i * 16 + (lane >> 4) * 8 + r;
        C[(long long)row * ldc + col] = acc[i][j][r] + bv;
      }
    }
  }
}

// =====================================================================
// Channel projection + LayerNorm + ReLU^2  (one block per (b,w) row)
// x4: [B,CH,BINS,W] ; out: [B*W, BINS] f32 + bf16 mirror
// =====================================================================
__global__ __launch_bounds__(256)
void k_inproj(const float* __restrict__ x4, const float* __restrict__ wch,
              const float* __restrict__ g, const float* __restrict__ be,
              float* __restrict__ outf, __bf16* __restrict__ outb)
{
  const int w = blockIdx.x, b = blockIdx.y, tid = threadIdx.x;
  __shared__ float buf[nBINS];
  __shared__ float rs[256], rq[256];
  float ws[nCH];
#pragma unroll
  for (int c = 0; c < nCH; ++c) ws[c] = wch[c];

  float lsum = 0.f, lsq = 0.f;
#pragma unroll
  for (int s = 0; s < nBINS / 256; ++s) {
    const int f = tid + s * 256;
    float a = 0.f;
#pragma unroll
    for (int c = 0; c < nCH; ++c)
      a += ws[c] * x4[(((long long)b * nCH + c) * nBINS + f) * nW + w];
    buf[f] = a;
    lsum += a;
    lsq  += a * a;
  }
  rs[tid] = lsum; rq[tid] = lsq;
  __syncthreads();
  for (int s = 128; s > 0; s >>= 1) {
    if (tid < s) { rs[tid] += rs[tid + s]; rq[tid] += rq[tid + s]; }
    __syncthreads();
  }
  const float mean = rs[0] * (1.f / nBINS);
  const float var  = rq[0] * (1.f / nBINS) - mean * mean;
  const float rstd = rsqrtf(var + 1e-5f);
  const long long row = (long long)b * nW + w;
#pragma unroll
  for (int s = 0; s < nBINS / 256; ++s) {
    const int f = tid + s * 256;
    float v = (buf[f] - mean) * rstd * g[f] + be[f];
    v = fmaxf(v, 0.f);
    v = v * v;
    outf[row * nBINS + f] = v;
    outb[row * nBINS + f] = (__bf16)v;
  }
}

// =====================================================================
// y = LN(a + b1? + b2?) ; writes f32 (+ optional bf16 mirror). One block/row.
// =====================================================================
__global__ __launch_bounds__(256)
void k_resln(const float* __restrict__ a, const float* __restrict__ b1,
             const float* __restrict__ b2,
             const float* __restrict__ g, const float* __restrict__ be,
             float* __restrict__ outf, __bf16* __restrict__ outb, int C)
{
  const long long row = blockIdx.x;
  const int tid = threadIdx.x;
  const float* pa = a + row * C;
  const float* p1 = b1 ? b1 + row * C : nullptr;
  const float* p2 = b2 ? b2 + row * C : nullptr;
  __shared__ float rs[256], rq[256];
  float s = 0.f, q = 0.f;
  for (int i = tid; i < C; i += 256) {
    float v = pa[i];
    if (p1) v += p1[i];
    if (p2) v += p2[i];
    s += v; q += v * v;
  }
  rs[tid] = s; rq[tid] = q;
  __syncthreads();
  for (int st = 128; st > 0; st >>= 1) {
    if (tid < st) { rs[tid] += rs[tid + st]; rq[tid] += rq[tid + st]; }
    __syncthreads();
  }
  const float mean = rs[0] / C;
  const float var  = rq[0] / C - mean * mean;
  const float rstd = rsqrtf(var + 1e-5f);
  for (int i = tid; i < C; i += 256) {
    float v = pa[i];
    if (p1) v += p1[i];
    if (p2) v += p2[i];
    const float y = (v - mean) * rstd * g[i] + be[i];
    outf[row * C + i] = y;
    if (outb) outb[row * C + i] = (__bf16)y;
  }
}

// ---------------- depthwise conv along t (x,y: [Bn,T,Cc]; w: [Cc*Kk]) -------
__global__ __launch_bounds__(256)
void k_dwconv(const float* __restrict__ x, const float* __restrict__ w,
              const float* __restrict__ bias, float* __restrict__ y,
              int Bn, int T, int Cc, int Kk, int pad, long long n)
{
  const long long idx = (long long)blockIdx.x * 256 + threadIdx.x;
  if (idx >= n) return;
  const int c = (int)(idx % Cc);
  const int t = (int)((idx / Cc) % T);
  const int b = (int)(idx / ((long long)Cc * T));
  float sv = bias ? bias[c] : 0.f;
  for (int j = 0; j < Kk; ++j) {
    const int tt = t + j - pad;
    if (tt >= 0 && tt < T)
      sv += x[((long long)b * T + tt) * Cc + c] * w[c * Kk + j];
  }
  y[idx] = sv;
}

// ---------------- softmax row with rel^T skew ----------------
__global__ __launch_bounds__(256)
void k_softmax(const float* __restrict__ qk, const float* __restrict__ rel,
               __bf16* __restrict__ probs)
{
  const int i = blockIdx.x;                 // query position (0..nW-1)
  const int z = blockIdx.y;                 // b*H + h
  const int tid = threadIdx.x;
  const float* qrow = qk  + ((long long)z * nW + i) * nW;
  const float* relb = rel + (long long)z * nW * nW;
  __shared__ float red[256];
  const float scale = 1.0f / 32.0f;         // 1/sqrt(BINS)

  float s0[2];
  float mx = -1e30f;
#pragma unroll
  for (int e = 0; e < 2; ++e) {
    const int j = tid + e * 256;
    const float v = (qrow[j] + relb[(long long)j * nW + i]) * scale;
    s0[e] = v;
    mx = fmaxf(mx, v);
  }
  red[tid] = mx; __syncthreads();
  for (int st = 128; st > 0; st >>= 1) {
    if (tid < st) red[tid] = fmaxf(red[tid], red[tid + st]);
    __syncthreads();
  }
  const float m = red[0];
  __syncthreads();
  float sum = 0.f;
#pragma unroll
  for (int e = 0; e < 2; ++e) { s0[e] = __expf(s0[e] - m); sum += s0[e]; }
  red[tid] = sum; __syncthreads();
  for (int st = 128; st > 0; st >>= 1) {
    if (tid < st) red[tid] += red[tid + st];
    __syncthreads();
  }
  const float inv = 1.0f / red[0];
  __bf16* prow = probs + ((long long)z * nW + i) * nW;
#pragma unroll
  for (int e = 0; e < 2; ++e) prow[tid + e * 256] = (__bf16)(s0[e] * inv);
}

// ---------------- small elementwise kernels ----------------
__global__ __launch_bounds__(256)
void k_cvt(const float* __restrict__ s, __bf16* __restrict__ d, long long n)
{
  const long long i = (long long)blockIdx.x * 256 + threadIdx.x;
  if (i < n) d[i] = (__bf16)s[i];
}
// f32 [R,C] -> bf16 [C,R]
__global__ __launch_bounds__(256)
void k_trans_f2b(const float* __restrict__ s, __bf16* __restrict__ d,
                 int R, int C, long long n)
{
  const long long i = (long long)blockIdx.x * 256 + threadIdx.x;
  if (i >= n) return;
  const int r = (int)(i % R);
  const int c = (int)(i / R);
  d[i] = (__bf16)s[(long long)r * C + c];
}
__global__ __launch_bounds__(256)
void k_relu2(float* __restrict__ x, long long n)
{
  const long long i = (long long)blockIdx.x * 256 + threadIdx.x;
  if (i < n) { float v = fmaxf(x[i], 0.f); x[i] = v * v; }
}
__global__ __launch_bounds__(256)
void k_silu(float* __restrict__ x, long long n)
{
  const long long i = (long long)blockIdx.x * 256 + threadIdx.x;
  if (i < n) { const float v = x[i]; x[i] = v / (1.f + __expf(-v)); }
}
__global__ __launch_bounds__(256)
void k_zero(float* __restrict__ x, long long n)
{
  const long long i = (long long)blockIdx.x * 256 + threadIdx.x;
  if (i < n) x[i] = 0.f;
}
// [B,W,H*D] f32 -> [B,H,W,D] bf16
__global__ __launch_bounds__(256)
void k_split(const float* __restrict__ s, __bf16* __restrict__ d, long long n)
{
  const long long i = (long long)blockIdx.x * 256 + threadIdx.x;
  if (i >= n) return;
  const int dd = (int)(i % nD);
  const int w  = (int)((i / nD) % nW);
  const int h  = (int)((i / ((long long)nD * nW)) % nH);
  const int b  = (int)(i / ((long long)nD * nW * nH));
  d[i] = (__bf16)s[((long long)b * nW + w) * nBINS + h * nD + dd];
}
// [B,W,H*D] f32 -> [B,H,D,W] bf16  (transposed heads for NT a@v GEMM)
__global__ __launch_bounds__(256)
void k_splitT(const float* __restrict__ s, __bf16* __restrict__ d, long long n)
{
  const long long i = (long long)blockIdx.x * 256 + threadIdx.x;
  if (i >= n) return;
  const int w  = (int)(i % nW);
  const int dd = (int)((i / nW) % nD);
  const int h  = (int)((i / ((long long)nW * nD)) % nH);
  const int b  = (int)(i / ((long long)nW * nD * nH));
  d[i] = (__bf16)s[((long long)b * nW + w) * nBINS + h * nD + dd];
}
// [B,H,W,D] f32 -> [B,W,H*D] bf16
__global__ __launch_bounds__(256)
void k_merge(const float* __restrict__ s, __bf16* __restrict__ d, long long n)
{
  const long long i = (long long)blockIdx.x * 256 + threadIdx.x;
  if (i >= n) return;
  const int c = (int)(i % nBINS);
  const int w = (int)((i / nBINS) % nW);
  const int b = (int)(i / ((long long)nBINS * nW));
  const int h = c / nD, dd = c % nD;
  d[i] = (__bf16)s[(((long long)b * nH + h) * nW + w) * nD + dd];
}
// [B,W,BINS] -> [B,1,BINS,W]
__global__ __launch_bounds__(256)
void k_out(const float* __restrict__ s, float* __restrict__ d, long long n)
{
  const long long i = (long long)blockIdx.x * 256 + threadIdx.x;
  if (i >= n) return;
  const int w = (int)(i % nW);
  const int f = (int)((i / nW) % nBINS);
  const int b = (int)(i / ((long long)nW * nBINS));
  d[i] = s[((long long)b * nW + w) * nBINS + f];
}

// =====================================================================
// host orchestration
// =====================================================================
static inline unsigned gb(long long n) { return (unsigned)((n + 255) / 256); }

extern "C" void kernel_launch(void* const* d_in, const int* in_sizes, int n_in,
                              void* d_out, int out_size, void* d_ws, size_t ws_size,
                              hipStream_t stream)
{
  (void)in_sizes; (void)n_in; (void)out_size; (void)ws_size;
  auto Pf = [&](int i) { return (const float*)d_in[i]; };

  // ---- workspace arena ----
  char* base = (char*)d_ws;
  size_t off = 0;
  auto aF = [&](long long n) { float* p = (float*)(base + off); off += ((size_t)n * 4 + 255) & ~(size_t)255; return p; };
  auto aH = [&](long long n) { __bf16* p = (__bf16*)(base + off); off += ((size_t)n * 2 + 255) & ~(size_t)255; return p; };

  float* A0f = aF((long long)nBW * nBINS);          // x activations (f32)
  float* A1f = aF((long long)nBW * nBINS);          // m activations (f32)
  float* T0f = aF((long long)nBW * nBINS);
  float* T1f = aF((long long)nBW * nBINS);
  float* T2f = aF((long long)nBW * nBINS);
  float* S0f = aF((long long)nB * nH * nW * nW);    // rel
  float* S1f = aF((long long)nB * nH * nW * nW);    // qk
  float* H0f = aF((long long)nBW * nC2);
  float* H1f = aF((long long)nBW * nC2);
  __bf16* Xb = aH((long long)nBW * nBINS);
  __bf16* Mb = aH((long long)nBW * nBINS);
  __bf16* Qh = aH((long long)nB * nH * nW * nD);    // also generic [BW,1024] bf16
  __bf16* Kh = aH((long long)nB * nH * nW * nD);
  __bf16* Vh = aH((long long)nB * nH * nW * nD);    // [B,H,D,W] (transposed)
  __bf16* Pb = aH((long long)nB * nH * nW * nW);
  __bf16* Gb = aH((long long)nBW * nC2);
  __bf16* Wb = aH((long long)nC2 * nBINS);          // weight staging (max 4096x1024)

  auto cvt = [&](const float* s, __bf16* d, long long n) {
    k_cvt<<<gb(n), 256, 0, stream>>>(s, d, n);
  };
  auto gemm = [&](const __bf16* A, long long sA, int lda,
                  const __bf16* Bm, long long sB, int ldb,
                  float* C, long long sC, int ldc, const float* bias,
                  int M, int N, int K, int batch) {
    dim3 g((unsigned)(N / GEMM_BN), (unsigned)(M / GEMM_BM), (unsigned)batch);
    k_gemm<<<g, 256, 0, stream>>>(A, sA, lda, Bm, sB, ldb, C, sC, ldc, bias, M, N, K);
  };
  auto gemmW = [&](const __bf16* A, const float* Wf, const float* bias,
                   float* C, int M, int N, int K) {
    cvt(Wf, Wb, (long long)N * K);
    gemm(A, 0, K, Wb, 0, K, C, 0, N, bias, M, N, K, 1);
  };
  auto dwc = [&](const float* x, const float* w, const float* bias, float* y,
                 int Cc, int Kk, int pad) {
    const long long n = (long long)nB * nW * Cc;
    k_dwconv<<<gb(n), 256, 0, stream>>>(x, w, bias, y, nB, nW, Cc, Kk, pad, n);
  };

  // one multi-head attention block; outT = [BW, BINS] f32
  auto attn = [&](const __bf16* xb, const __bf16* mb, int pb, float* outT) {
    const long long nHD = (long long)nB * nH * nW * nD;
    // q / k / v : GEMM + bias -> dwconv(k=3) + bias -> split heads (bf16)
    gemmW(xb, Pf(pb + 0), Pf(pb + 1), T0f, nBW, nBINS, nBINS);
    dwc(T0f, Pf(pb + 2), Pf(pb + 3), T1f, nBINS, 3, 1);
    k_split<<<gb(nHD), 256, 0, stream>>>(T1f, Qh, nHD);
    gemmW(mb, Pf(pb + 4), Pf(pb + 5), T0f, nBW, nBINS, nBINS);
    dwc(T0f, Pf(pb + 6), Pf(pb + 7), T1f, nBINS, 3, 1);
    k_split<<<gb(nHD), 256, 0, stream>>>(T1f, Kh, nHD);
    gemmW(mb, Pf(pb + 8), Pf(pb + 9), T0f, nBW, nBINS, nBINS);
    dwc(T0f, Pf(pb + 10), Pf(pb + 11), T1f, nBINS, 3, 1);
    k_splitT<<<gb(nHD), 256, 0, stream>>>(T1f, Vh, nHD);     // [B,H,D,W]
    // rel = q @ er : er [D,W] f32 -> er^T [W,D] bf16, NT GEMM
    k_trans_f2b<<<gb((long long)nD * nW), 256, 0, stream>>>(Pf(pb + 14), Wb,
                                                            nD, nW, (long long)nD * nW);
    gemm(Qh, (long long)nW * nD, nD, Wb, 0, nD,
         S0f, (long long)nW * nW, nW, nullptr, nW, nW, nD, nB * nH);
    // qk = q @ k^T (NT)
    gemm(Qh, (long long)nW * nD, nD, Kh, (long long)nW * nD, nD,
         S1f, (long long)nW * nW, nW, nullptr, nW, nW, nD, nB * nH);
    // softmax((qk + rel^T)/sqrt(BINS)) -> bf16
    k_softmax<<<dim3(nW, nB * nH), 256, 0, stream>>>(S1f, S0f, Pb);
    // o = a @ v : v stored [D,W] per head -> NT GEMM
    gemm(Pb, (long long)nW * nW, nW, Vh, (long long)nD * nW, nW,
         T0f, (long long)nW * nD, nD, nullptr, nW, nD, nW, nB * nH);
    // merge heads -> bf16, output projection + bias
    k_merge<<<gb((long long)nBW * nBINS), 256, 0, stream>>>(T0f, Qh, (long long)nBW * nBINS);
    gemmW(Qh, Pf(pb + 12), Pf(pb + 13), outT, nBW, nBINS, nBINS);
  };

  // ---- 1. input / memory projection + LN + ReLU^2 ----
  k_inproj<<<dim3(nW, nB), 256, 0, stream>>>((const float*)d_in[0], Pf(2), Pf(4), Pf(5), A0f, Xb);
  k_inproj<<<dim3(nW, nB), 256, 0, stream>>>((const float*)d_in[1], Pf(3), Pf(6), Pf(7), A1f, Mb);

  // ---- 2. sa1 + ea1 + LN(n1) ----
  attn(Xb, Xb, 20, T2f);
  attn(Xb, Mb, 35, H0f);
  k_resln<<<nBW, 256, 0, stream>>>(A0f, T2f, H0f, Pf(8), Pf(9), A0f, Xb, nBINS);

  // ---- 3. conv-gated FFN ----
  dwc(A0f, Pf(80), nullptr, T0f, nBINS, 11, 5);
  cvt(T0f, Qh, (long long)nBW * nBINS);
  gemmW(Qh, Pf(81), nullptr, H0f, nBW, nC2, nBINS);
  k_relu2<<<gb((long long)nBW * nC2), 256, 0, stream>>>(H0f, (long long)nBW * nC2);
  dwc(A0f, Pf(82), nullptr, T0f, nBINS, 7, 3);
  cvt(T0f, Qh, (long long)nBW * nBINS);
  k_zero<<<gb((long long)nBW * nC2), 256, 0, stream>>>(H1f, (long long)nBW * nC2);
  cvt(Pf(83), Wb, (long long)(nFF / 2) * nBINS);
  gemm(Qh, 0, nBINS, Wb, 0, nBINS, H1f, 0, nC2, nullptr, nBW, nFF / 2, nBINS, 1);
  k_resln<<<nBW, 256, 0, stream>>>(H0f, H1f, nullptr, Pf(18), Pf(19), H0f, (__bf16*)nullptr, nC2);
  dwc(H0f, Pf(84), nullptr, H1f, nC2, 7, 3);
  cvt(H1f, Gb, (long long)nBW * nC2);
  gemmW(Gb, Pf(85), nullptr, T0f, nBW, nBINS, nC2);
  k_resln<<<nBW, 256, 0, stream>>>(A0f, T0f, nullptr, Pf(10), Pf(11), A0f, Xb, nBINS);  // n3

  // ---- 4. sa2 / ea2 ----
  attn(Xb, Xb, 50, T2f);
  k_resln<<<nBW, 256, 0, stream>>>(A0f, T2f, nullptr, Pf(12), Pf(13), A0f, Xb, nBINS);  // n4
  attn(Xb, Mb, 65, T2f);
  k_resln<<<nBW, 256, 0, stream>>>(A0f, T2f, nullptr, Pf(14), Pf(15), A0f, Xb, nBINS);  // n5

  // ---- 5. final SiLU FFN + LN(n6) ----
  gemmW(Xb, Pf(86), nullptr, H0f, nBW, nC2, nBINS);
  k_silu<<<gb((long long)nBW * nC2), 256, 0, stream>>>(H0f, (long long)nBW * nC2);
  cvt(H0f, Gb, (long long)nBW * nC2);
  gemmW(Gb, Pf(87), nullptr, T0f, nBW, nBINS, nC2);
  k_resln<<<nBW, 256, 0, stream>>>(A0f, T0f, nullptr, Pf(16), Pf(17), A0f, (__bf16*)nullptr, nBINS);

  // ---- 6. transpose to [B,1,BINS,W] ----
  const long long no = (long long)nB * nBINS * nW;
  k_out<<<gb(no), 256, 0, stream>>>(A0f, (float*)d_out, no);
}
